// GrammarInference_21766894256718
// MI455X (gfx1250) — matrix operand order repaired and checked by reading, not compiled, not run on hardware
//
#include <hip/hip_runtime.h>
#include <hip/hip_bf16.h>

typedef __attribute__((ext_vector_type(16))) _Float16 v16h;
typedef __attribute__((ext_vector_type(8)))  _Float16 v8h;
typedef __attribute__((ext_vector_type(8)))  float    v8f;
typedef __attribute__((ext_vector_type(4)))  float    v4f;

#define DEVINL __device__ __forceinline__

// ---- model constants ----
constexpr int CB  = 2048;   // batch
constexpr int CP  = 128;    // program length
constexpr int CT  = 64;     // trace length
constexpr int CV  = 1000;   // vocab
constexpr int CNC = 32;     // n choice names
constexpr int CNO = 16;     // options per choice
constexpr int CE  = 300;    // embedding size
constexpr int CH  = 256;    // hidden
constexpr int CG3 = 768;    // 3*H
constexpr int CGIN= 856;    // GRU input size
constexpr int BT  = 16;     // batch rows per workgroup
constexpr int KC  = CH/32;  // 8 k-chunks of 32 over H

// ---------------------------------------------------------------------------
// WMMA helper: D = A(16x32 f16) x B(32x16 f16) + C (f32)
// ---------------------------------------------------------------------------
DEVINL v8f wmma_f16(v16h a, v16h b, v8f c) {
  return __builtin_amdgcn_wmma_f32_16x16x32_f16(
      /*neg_a=*/false, a, /*neg_b=*/false, b,
      /*c_mod=*/(short)0, c, /*reuse_a=*/false, /*reuse_b=*/false);
}

// A fragment (16x32 f16) from an LDS tile of h laid out [BT][CH] row-major.
// ISA layout: lanes 0-15 hold M=lane, K j<8 -> k0+j, j>=8 -> k0+16+(j-8);
//             lanes 16-31 hold M=lane-16, k offsets +8 / +24.
DEVINL v16h load_afrag(const _Float16* hbase, int ldh, int lane, int k0) {
  int r   = lane & 15;
  int sel = (lane >> 4) & 1;
  const _Float16* p = hbase + r * ldh;
  v8h lo = *(const v8h*)(p + k0 + sel * 8);
  v8h hi = *(const v8h*)(p + k0 + 16 + sel * 8);
  v16h a;
#pragma unroll
  for (int j = 0; j < 8; ++j) { a[j] = lo[j]; a[8 + j] = hi[j]; }
  return a;
}

// B fragment pre-swizzled in global memory: frags[fragIdx][lane][16 halves]
DEVINL v16h load_bfrag(const _Float16* frags, int fragIdx, int lane) {
  const _Float16* p = frags + ((size_t)fragIdx * 32 + lane) * 16;
  v8h lo = *(const v8h*)(p);
  v8h hi = *(const v8h*)(p + 8);
  v16h b;
#pragma unroll
  for (int j = 0; j < 8; ++j) { b[j] = lo[j]; b[8 + j] = hi[j]; }
  return b;
}

// Branch-free activations on hardware transcendentals (v_exp_f32 + v_rcp_f32).
// No EXEC divergence inside the WMMA region; saturation via inf propagation.
DEVINL float sigmoid_fast(float x) {
  return __builtin_amdgcn_rcpf(1.0f + __expf(-x));
}
DEVINL float tanh_fast(float x) {
  return 1.0f - 2.0f * __builtin_amdgcn_rcpf(1.0f + __expf(2.0f * x));
}

// ---------------------------------------------------------------------------
// Prep: swizzle W[N][K] (f32, row-major, row stride ldk) into WMMA B
// fragments (f16).  B[k][n] = W[n][k].
// Layout: lane<16 -> N=n0+lane, K j<8 -> k0+j, j>=8 -> k0+16+(j-8); lane>=16 -> N=n0+lane-16, k offsets +8 / +24.
// ---------------------------------------------------------------------------
__global__ void prep_frag_kernel(const float* __restrict__ W, _Float16* __restrict__ out,
                                 int ntiles, int kchunks, int ldk) {
  int idx = blockIdx.x * blockDim.x + threadIdx.x;   // one thread per (frag,lane)
  int total = ntiles * kchunks * 32;
  if (idx >= total) return;
  int lane = idx & 31;
  int fi   = idx >> 5;
  int kc   = fi % kchunks;
  int nt   = fi / kchunks;
  int n     = nt * 16 + (lane & 15);
  int kbase = kc * 32 + ((lane >> 4) ? 8 : 0);
  const float* w = W + (size_t)n * ldk + kbase;
  _Float16* o = out + (size_t)idx * 16;
#pragma unroll
  for (int j = 0; j < 16; ++j) o[j] = (_Float16)w[j < 8 ? j : j + 8];
}

// embW[v][n] = sum_e token_emb[v][e]*Wih_enc[n][e] + bih[n] + bhh[n]
__global__ void prep_embW_kernel(const float* __restrict__ tok, const float* __restrict__ Wih,
                                 const float* __restrict__ bih, const float* __restrict__ bhh,
                                 float* __restrict__ out) {
  int idx = blockIdx.x * blockDim.x + threadIdx.x;
  if (idx >= CV * CH) return;
  int n = idx & (CH - 1);
  int v = idx >> 8;
  const float* t = tok + (size_t)v * CE;
  const float* w = Wih + (size_t)n * CE;
  float acc = bih[n] + bhh[n];
  for (int e = 0; e < CE; ++e) acc += t[e] * w[e];
  out[idx] = acc;
}

// chW[nc][no][g] = sum_e choice_tab[nc][no][e]*Wih_gru[g][e]
__global__ void prep_chW_kernel(const float* __restrict__ ctab, const float* __restrict__ Wih,
                                float* __restrict__ out) {
  int idx = blockIdx.x * blockDim.x + threadIdx.x;
  if (idx >= CNC * CNO * CG3) return;
  int g    = idx % CG3;
  int pair = idx / CG3;                 // nc*16+no
  const float* c = ctab + (size_t)pair * CE;
  const float* w = Wih + (size_t)g * CGIN;   // columns [0,300)
  float acc = 0.f;
  for (int e = 0; e < CE; ++e) acc += c[e] * w[e];
  out[idx] = acc;
}

// idxW[nc][g] = sum_e index_tab[nc][e]*Wih_gru[g][556+e]
__global__ void prep_idxW_kernel(const float* __restrict__ itab, const float* __restrict__ Wih,
                                 float* __restrict__ out) {
  int idx = blockIdx.x * blockDim.x + threadIdx.x;
  if (idx >= CNC * CG3) return;
  int g  = idx % CG3;
  int nc = idx / CG3;
  const float* it = itab + (size_t)nc * CE;
  const float* w  = Wih + (size_t)g * CGIN + (CE + CH);  // columns [556,856)
  float acc = 0.f;
  for (int e = 0; e < CE; ++e) acc += it[e] * w[e];
  out[idx] = acc;
}

// ---------------------------------------------------------------------------
// Persistent encoder: tanh-RNN, 128 steps, h kept in LDS as f16.
// Each WG: 16 batch rows, 8 waves; wave w computes hidden col-tiles {w, w+8}.
// Epilogue: gx_base tile = final_h @ W2^T + bih_gru, also via WMMA.
// ---------------------------------------------------------------------------
__global__ __launch_bounds__(256) void encoder_kernel(
    const int* __restrict__ program, const int* __restrict__ program_len,
    const float* __restrict__ h0, const float* __restrict__ embW,
    const _Float16* __restrict__ fragWhh, const _Float16* __restrict__ fragW2,
    const float* __restrict__ bih_gru, float* __restrict__ gxBase) {
  __shared__ _Float16 lds_h[BT][CH];   // 8 KB
  __shared__ float    lds_x[BT][CH];   // 16 KB
  __shared__ int      tokBuf[BT];
  __shared__ int      lenBuf[BT];

  const int tid   = threadIdx.x;
  const int lane  = tid & 31;
  const int wave  = tid >> 5;
  const int bbase = blockIdx.x * BT;

  for (int i = tid; i < BT * CH; i += 256) {
    int m = i >> 8, n = i & (CH - 1);
    lds_h[m][n] = (_Float16)h0[(size_t)(bbase + m) * CH + n];
  }
  if (tid < BT) lenBuf[tid] = program_len[bbase + tid];
  __syncthreads();

  const int nt0 = wave, nt1 = wave + 8;
  const int col  = lane & 15;
  const int rsel = (lane >> 4) ? 8 : 0;

  for (int t = 0; t < CP; ++t) {
    if (tid < BT) tokBuf[tid] = program[(size_t)(bbase + tid) * CP + t];
    __syncthreads();

    // stage x_add = embW[token] (coalesced v4 loads)
    for (int i = tid; i < BT * (CH / 4); i += 256) {
      int m = i / (CH / 4), c4 = i % (CH / 4);
      v4f v = *(const v4f*)(embW + (size_t)tokBuf[m] * CH + c4 * 4);
      *(v4f*)&lds_x[m][c4 * 4] = v;
    }
    // A fragments from current h (lds_h is stable until next step's writes)
    v16h afr[KC];
#pragma unroll
    for (int kc = 0; kc < KC; ++kc) afr[kc] = load_afrag(&lds_h[0][0], CH, lane, kc * 32);
    __syncthreads();

    v8f acc0 = {}, acc1 = {};
#pragma unroll
    for (int kc = 0; kc < KC; ++kc) {
      v16h b0 = load_bfrag(fragWhh, nt0 * KC + kc, lane);
      v16h b1 = load_bfrag(fragWhh, nt1 * KC + kc, lane);
      acc0 = wmma_f16(afr[kc], b0, acc0);
      acc1 = wmma_f16(afr[kc], b1, acc1);
    }
    // h_new = tanh(x_add + h@Whh^T); masked by t < len
#pragma unroll
    for (int r = 0; r < 8; ++r) {
      int m = r + rsel;
      bool valid = (t < lenBuf[m]);
      {
        int n = nt0 * 16 + col;
        float hnew = tanh_fast(acc0[r] + lds_x[m][n]);
        float hold = (float)lds_h[m][n];
        lds_h[m][n] = (_Float16)(valid ? hnew : hold);
      }
      {
        int n = nt1 * 16 + col;
        float hnew = tanh_fast(acc1[r] + lds_x[m][n]);
        float hold = (float)lds_h[m][n];
        lds_h[m][n] = (_Float16)(valid ? hnew : hold);
      }
    }
    // next-iteration barrier (after tokBuf staging) orders these writes
  }
  __syncthreads();

  // Epilogue: gx_base[b][g] = prog_h[b] @ W2^T[.,g] + bih_gru[g]  (48 WMMAs)
  {
    v16h afr[KC];
#pragma unroll
    for (int kc = 0; kc < KC; ++kc) afr[kc] = load_afrag(&lds_h[0][0], CH, lane, kc * 32);
#pragma unroll
    for (int q = 0; q < 6; ++q) {
      int nt = wave + q * 8;          // 48 n-tiles over 3H, 6 per wave
      v8f acc = {};
#pragma unroll
      for (int kc = 0; kc < KC; ++kc) {
        v16h b = load_bfrag(fragW2, nt * KC + kc, lane);
        acc = wmma_f16(afr[kc], b, acc);
      }
      int n = nt * 16 + col;
      float bv = bih_gru[n];
#pragma unroll
      for (int r = 0; r < 8; ++r) {
        int m = r + rsel;
        gxBase[(size_t)(bbase + m) * CG3 + n] = acc[r] + bv;
      }
    }
  }
}

// ---------------------------------------------------------------------------
// Persistent GRU decoder + per-step inference head.
// gh = h@Whh_gru^T via WMMA; gx gathered from precomputed tables.
// ---------------------------------------------------------------------------
__global__ __launch_bounds__(256) void decoder_kernel(
    const int* __restrict__ trace, const int* __restrict__ choices,
    const float* __restrict__ h0, const float* __restrict__ gxBase,
    const float* __restrict__ chW, const float* __restrict__ idxW,
    const _Float16* __restrict__ fragWhh, const float* __restrict__ bhh,
    const float* __restrict__ infW, const float* __restrict__ infb,
    float* __restrict__ out /* [T-1][B][NO] */) {
  __shared__ _Float16 lds_h[BT][CH];     // 8 KB
  __shared__ float    lds_gx[BT][CG3];   // 48 KB
  __shared__ float    lds_bhh[CG3];      // 3 KB
  __shared__ int      prevBuf[BT], optBuf[BT], curBuf[BT];

  const int tid   = threadIdx.x;
  const int lane  = tid & 31;
  const int wave  = tid >> 5;
  const int bbase = blockIdx.x * BT;

  for (int i = tid; i < BT * CH; i += 256) {
    int m = i >> 8, n = i & (CH - 1);
    lds_h[m][n] = (_Float16)h0[(size_t)(bbase + m) * CH + n];
  }
  for (int i = tid; i < CG3; i += 256) lds_bhh[i] = bhh[i];
  __syncthreads();

  const int ct0 = wave, ct1 = wave + 8;     // output hidden col-tiles
  const int col  = lane & 15;
  const int rsel = (lane >> 4) ? 8 : 0;

  for (int t = 0; t < CT - 1; ++t) {
    if (tid < BT) {
      int b  = bbase + tid;
      int pv = trace[(size_t)b * CT + t];
      prevBuf[tid] = pv;
      curBuf[tid]  = trace[(size_t)b * CT + t + 1];
      optBuf[tid]  = choices[(size_t)b * CNC + pv];
    }
    __syncthreads();

    // stage gx = gx_base + chW[prev,opt] + idxW[prev]   (16 x 768 f32)
    for (int i = tid; i < BT * (CG3 / 4); i += 256) {
      int m = i / (CG3 / 4), c4 = i % (CG3 / 4);
      int pv = prevBuf[m];
      v4f a = *(const v4f*)(gxBase + (size_t)(bbase + m) * CG3 + c4 * 4);
      v4f c = *(const v4f*)(chW + ((size_t)(pv * CNO + optBuf[m])) * CG3 + c4 * 4);
      v4f d = *(const v4f*)(idxW + (size_t)pv * CG3 + c4 * 4);
      *(v4f*)&lds_gx[m][c4 * 4] = a + c + d;
    }
    v16h afr[KC];
#pragma unroll
    for (int kc = 0; kc < KC; ++kc) afr[kc] = load_afrag(&lds_h[0][0], CH, lane, kc * 32);
    __syncthreads();

#pragma unroll
    for (int ti = 0; ti < 2; ++ti) {
      int ct = (ti == 0) ? ct0 : ct1;
      v8f aR = {}, aZ = {}, aN = {};
#pragma unroll
      for (int kc = 0; kc < KC; ++kc) {
        v16h bR = load_bfrag(fragWhh, (ct +  0) * KC + kc, lane);
        v16h bZ = load_bfrag(fragWhh, (ct + 16) * KC + kc, lane);
        v16h bN = load_bfrag(fragWhh, (ct + 32) * KC + kc, lane);
        aR = wmma_f16(afr[kc], bR, aR);
        aZ = wmma_f16(afr[kc], bZ, aZ);
        aN = wmma_f16(afr[kc], bN, aN);
      }
      int n = ct * 16 + col;
      float bRv = lds_bhh[n], bZv = lds_bhh[CH + n], bNv = lds_bhh[2 * CH + n];
#pragma unroll
      for (int r = 0; r < 8; ++r) {
        int m = r + rsel;
        float hr = aR[r] + bRv, hz = aZ[r] + bZv, hn = aN[r] + bNv;
        float xr = lds_gx[m][n], xz = lds_gx[m][CH + n], xn = lds_gx[m][2 * CH + n];
        float rg = sigmoid_fast(xr + hr);
        float zg = sigmoid_fast(xz + hz);
        float ng = tanh_fast(xn + rg * hn);
        float hold = (float)lds_h[m][n];
        lds_h[m][n] = (_Float16)((1.0f - zg) * ng + zg * hold);
      }
    }
    __syncthreads();

    // inference head: pred[b][o] = h_new[b] . infW[cur][o] + infb[cur][o]
    {
      int row = 2 * wave + ((lane >> 4) & 1);
      int o   = lane & 15;
      int b   = bbase + row;
      int cur = curBuf[row];
      const float* W = infW + ((size_t)cur * CNO + o) * CH;
      float acc = infb[(size_t)cur * CNO + o];
#pragma unroll 4
      for (int hh = 0; hh < CH; hh += 8) {
        v8h hv = *(const v8h*)&lds_h[row][hh];
        v4f w0 = *(const v4f*)(W + hh);
        v4f w1 = *(const v4f*)(W + hh + 4);
        acc += (float)hv[0] * w0[0] + (float)hv[1] * w0[1] +
               (float)hv[2] * w0[2] + (float)hv[3] * w0[3] +
               (float)hv[4] * w1[0] + (float)hv[5] * w1[1] +
               (float)hv[6] * w1[2] + (float)hv[7] * w1[3];
      }
      out[((size_t)t * CB + b) * CNO + o] = acc;
    }
    __syncthreads();   // protect curBuf / lds_h for next step
  }
}

// ---------------------------------------------------------------------------
// Host-side launcher
// ---------------------------------------------------------------------------
static inline size_t align256(size_t x) { return (x + 255) & ~(size_t)255; }

extern "C" void kernel_launch(void* const* d_in, const int* in_sizes, int n_in,
                              void* d_out, int out_size, void* d_ws, size_t ws_size,
                              hipStream_t stream) {
  (void)in_sizes; (void)n_in; (void)out_size; (void)ws_size;

  const int*   program     = (const int*)d_in[0];
  const int*   program_len = (const int*)d_in[1];
  const int*   trace       = (const int*)d_in[2];
  // d_in[3] = trace_len (unused by the reference output)
  const int*   choices     = (const int*)d_in[4];
  const float* h0_enc      = (const float*)d_in[5];
  const float* h0_dec      = (const float*)d_in[6];
  const float* token_emb   = (const float*)d_in[7];
  const float* Wih_enc     = (const float*)d_in[8];
  const float* Whh_enc     = (const float*)d_in[9];
  const float* bih_enc     = (const float*)d_in[10];
  const float* bhh_enc     = (const float*)d_in[11];
  const float* choice_tab  = (const float*)d_in[12];
  const float* index_tab   = (const float*)d_in[13];
  const float* Wih_gru     = (const float*)d_in[14];
  const float* Whh_gru     = (const float*)d_in[15];
  const float* bih_gru     = (const float*)d_in[16];
  const float* bhh_gru     = (const float*)d_in[17];
  const float* infW        = (const float*)d_in[18];
  const float* infb        = (const float*)d_in[19];
  float*       out         = (float*)d_out;

  // workspace carve-up
  char* ws = (char*)d_ws;
  size_t o = 0;
  _Float16* fragEnc = (_Float16*)(ws + o); o = align256(o + (size_t)16 * KC * 32 * 16 * sizeof(_Float16));
  _Float16* fragGru = (_Float16*)(ws + o); o = align256(o + (size_t)48 * KC * 32 * 16 * sizeof(_Float16));
  _Float16* fragW2  = (_Float16*)(ws + o); o = align256(o + (size_t)48 * KC * 32 * 16 * sizeof(_Float16));
  float*    embW    = (float*)(ws + o);    o = align256(o + (size_t)CV * CH * sizeof(float));
  float*    chW     = (float*)(ws + o);    o = align256(o + (size_t)CNC * CNO * CG3 * sizeof(float));
  float*    idxW    = (float*)(ws + o);    o = align256(o + (size_t)CNC * CG3 * sizeof(float));
  float*    gxBase  = (float*)(ws + o);    o = align256(o + (size_t)CB * CG3 * sizeof(float));

  // one-time weight prep (all ordered on `stream`)
  prep_frag_kernel<<<(16 * KC * 32 + 255) / 256, 256, 0, stream>>>(Whh_enc, fragEnc, 16, KC, CH);
  prep_frag_kernel<<<(48 * KC * 32 + 255) / 256, 256, 0, stream>>>(Whh_gru, fragGru, 48, KC, CH);
  // W2 = Wih_gru columns [300,556)  ->  B[k][g] = Wih_gru[g][300+k]
  prep_frag_kernel<<<(48 * KC * 32 + 255) / 256, 256, 0, stream>>>(Wih_gru + CE, fragW2, 48, KC, CGIN);
  prep_embW_kernel<<<(CV * CH + 255) / 256, 256, 0, stream>>>(token_emb, Wih_enc, bih_enc, bhh_enc, embW);
  prep_chW_kernel<<<(CNC * CNO * CG3 + 255) / 256, 256, 0, stream>>>(choice_tab, Wih_gru, chW);
  prep_idxW_kernel<<<(CNC * CG3 + 255) / 256, 256, 0, stream>>>(index_tab, Wih_gru, idxW);

  // encoder (persistent RNN over 128 steps) + fused gx_base WMMA epilogue
  encoder_kernel<<<CB / BT, 256, 0, stream>>>(program, program_len, h0_enc, embW,
                                              fragEnc, fragW2, bih_gru, gxBase);

  // decoder (persistent GRU over 63 steps, fused inference head)
  decoder_kernel<<<CB / BT, 256, 0, stream>>>(trace, choices, h0_dec, gxBase, chW, idxW,
                                              fragGru, bhh_gru, infW, infb, out);
}
